// LocalAttentionParallel_17317308138078
// MI455X (gfx1250) — compile-verified, matrix operand order checked by reading, not run on hardware
//
#include <hip/hip_runtime.h>
#include <hip/hip_bf16.h>

// ---------------------------------------------------------------------------
// MI455X (gfx1250) implementation: banded local attention block.
//   q/k/v = MLP(x) [bf16 WMMA GEMMs, fp32 accum]
//   S = band_mask(q k^T / sqrt(768*256))  (only the 320-wide window per
//       64-row query block is nonzero in the reference)
//   y = S v   -> LayerNorm -> out (fp32)
// Matmuls via v_wmma_f32_16x16x32_bf16; MLP GEMMs stage HBM->LDS with
// CDNA5 async copies (ASYNCcnt), double-buffered, one barrier per k-step.
// ---------------------------------------------------------------------------

#define N_EMBD 768
#define T_LEN  4096
#define BATCH  4
#define SPAN   256

typedef __attribute__((ext_vector_type(16))) __bf16 v16bf;
typedef __attribute__((ext_vector_type(8)))  float  v8f;
typedef int vint4 __attribute__((ext_vector_type(4)));

union FragU { vint4 i[2]; v16bf v; };   // 32 bytes; trivially copyable

__device__ __forceinline__ unsigned short f2bf(float f) {
  unsigned u = __builtin_bit_cast(unsigned, f);
  u += 0x7FFFu + ((u >> 16) & 1u);      // round-to-nearest-even
  return (unsigned short)(u >> 16);
}

// --------------------- CDNA5 async HBM->LDS copy (16B) ---------------------
#define AS1 __attribute__((address_space(1)))
#define AS3 __attribute__((address_space(3)))

__device__ __forceinline__ void async_copy16(const unsigned short* g,
                                             unsigned short* l) {
#if !defined(__HIP_DEVICE_COMPILE__)
  (void)g; (void)l;
#elif __has_builtin(__builtin_amdgcn_global_load_async_to_lds_b128)
  __builtin_amdgcn_global_load_async_to_lds_b128(
      (AS1 vint4*)g, (AS3 vint4*)l, 0, 0);
#else
  const unsigned lds = (unsigned)(unsigned long long)l;   // LDS byte address
  asm volatile("global_load_async_to_lds_b128 %0, %1, off"
               :: "v"(lds), "v"((unsigned long long)g) : "memory");
#endif
}

__device__ __forceinline__ void async_wait0() {
#if !defined(__HIP_DEVICE_COMPILE__)
#elif __has_builtin(__builtin_amdgcn_s_wait_asynccnt)
  __builtin_amdgcn_s_wait_asynccnt(0);
#else
  asm volatile("s_wait_asynccnt 0" ::: "memory");
#endif
}

// ---------------------------------------------------------------------------
// Blocked GEMM:  C(BM x BN) = A(BM x K) * Bt(BN x K)^T  (+bias, act, mask)
//   8 waves arranged WM x WN; each wave computes TM x TN 16x16 tiles.
//   BM = WM*TM*16, BN = WN*TN*16.  256 threads.
//   Bt logical row n -> global row brow0+n (zero outside [0,browsHi));
//   logical col kk -> global col bcol0+kk (zero outside [0,bcolsHi)).
//   ASYNC=1 (requires GUARD=0): HBM->LDS via async b128 copies, waited with
//   s_wait_asynccnt before the per-step barrier.  ASYNC=0: predicated VGPR
//   staging with zero-fill (band edges).
// ---------------------------------------------------------------------------
template <int WM, int WN, int TM, int TN,
          int ACT_GELU, int OUT_FP32, int MASK, int GUARD, int ASYNC>
__device__ __forceinline__ void gemm_block(
    const unsigned short* __restrict__ A, int lda,
    const unsigned short* __restrict__ Bt, int brow0, int ldb,
    int bcol0, unsigned browsHi, unsigned bcolsHi,
    const float* __restrict__ bias,
    void* __restrict__ Cp, int ldc, int K,
    int gi0, int gj0, float scale)
{
  constexpr int BM = WM * TM * 16;
  constexpr int BN = WN * TN * 16;
  constexpr int NA = BM / 64;              // vint4 per thread for A panel
  constexpr int NB = BN / 64;              // vint4 per thread for B panel
  constexpr int BUFI4 = (BM + BN) * 4;     // vint4 per LDS buffer

  __shared__ __align__(16) unsigned short smem[2 * (BM + BN) * 32];

  const int tid  = threadIdx.x;
  const int lane = tid & 31;
  const int wave = tid >> 5;
  const int wm   = wave / WN;
  const int wn   = wave % WN;
  const int half = lane >> 4;
  const int lr   = lane & 15;

  vint4 ra[NA], rb[NB];

  auto loadPanels = [&](int ks) {
#pragma unroll
    for (int i = 0; i < NA; ++i) {
      const int idx = tid + i * 256;
      const int row = idx >> 2, seg = idx & 3;
      ra[i] = *(const vint4*)(A + (long)row * lda + ks + seg * 8);
    }
#pragma unroll
    for (int i = 0; i < NB; ++i) {
      const int idx = tid + i * 256;
      const int row = idx >> 2, seg = idx & 3;
      if (GUARD) {
        const int grow = brow0 + row;
        const int gcol = bcol0 + ks + seg * 8;
        vint4 v = {0, 0, 0, 0};
        if ((unsigned)grow < browsHi && (unsigned)gcol < bcolsHi)
          v = *(const vint4*)(Bt + (long)grow * ldb + gcol);
        rb[i] = v;
      } else {
        rb[i] = *(const vint4*)(Bt + (long)(brow0 + row) * ldb + bcol0 + ks + seg * 8);
      }
    }
  };
  auto storePanels = [&](int buf) {
    vint4* dA = (vint4*)smem + buf * BUFI4;
    vint4* dB = dA + BM * 4;
#pragma unroll
    for (int i = 0; i < NA; ++i) dA[tid + i * 256] = ra[i];
#pragma unroll
    for (int i = 0; i < NB; ++i) dB[tid + i * 256] = rb[i];
  };
  auto issueAsync = [&](int ks, int buf) {
    unsigned short* base = smem + buf * (BM + BN) * 32;
#pragma unroll
    for (int i = 0; i < NA; ++i) {
      const int idx = tid + i * 256;
      const int row = idx >> 2, seg = idx & 3;
      async_copy16(A + (long)row * lda + ks + seg * 8, base + idx * 8);
    }
#pragma unroll
    for (int i = 0; i < NB; ++i) {
      const int idx = tid + i * 256;
      const int row = idx >> 2, seg = idx & 3;
      async_copy16(Bt + (long)(brow0 + row) * ldb + bcol0 + ks + seg * 8,
                   base + BM * 32 + idx * 8);
    }
  };

  v8f acc[TM][TN] = {};

  auto computeStep = [&](int cur) {
    const vint4* pA = (const vint4*)smem + cur * BUFI4;
    const vint4* pB = pA + BM * 4;
    FragU fa[TM], fb[TN];
#pragma unroll
    for (int mt = 0; mt < TM; ++mt) {
      const int r = (wm * TM + mt) * 16 + lr;
      fa[mt].i[0] = pA[r * 4 + half];           // K = half*8 + 0..7
      fa[mt].i[1] = pA[r * 4 + 2 + half];       // K = 16 + half*8 + 0..7
    }
#pragma unroll
    for (int nt = 0; nt < TN; ++nt) {
      const int r = (wn * TN + nt) * 16 + lr;
      fb[nt].i[0] = pB[r * 4 + half];
      fb[nt].i[1] = pB[r * 4 + 2 + half];
    }
#pragma unroll
    for (int mt = 0; mt < TM; ++mt)
#pragma unroll
      for (int nt = 0; nt < TN; ++nt)
        acc[mt][nt] = __builtin_amdgcn_wmma_f32_16x16x32_bf16(
            false, fa[mt].v, false, fb[nt].v, (short)0, acc[mt][nt],
            false, false);
  };

  const int ksteps = K / 32;
  if (ASYNC) {
    issueAsync(0, 0);
    async_wait0();
    __syncthreads();
    for (int s = 0; s < ksteps; ++s) {
      const int cur = s & 1;
      if (s + 1 < ksteps) issueAsync((s + 1) * 32, cur ^ 1);  // overlap WMMAs
      computeStep(cur);
      if (s + 1 < ksteps) async_wait0();
      __syncthreads();
    }
  } else {
    loadPanels(0);
    storePanels(0);
    __syncthreads();
    for (int s = 0; s < ksteps; ++s) {
      const int cur = s & 1;
      if (s + 1 < ksteps) loadPanels((s + 1) * 32);           // overlap WMMAs
      computeStep(cur);
      if (s + 1 < ksteps) storePanels(cur ^ 1);
      __syncthreads();
    }
  }

  // epilogue: C layout per ISA (VGPR r -> M = r + 8*half, N = lane%16)
#pragma unroll
  for (int mt = 0; mt < TM; ++mt)
#pragma unroll
    for (int nt = 0; nt < TN; ++nt) {
      const int nlocal = (wn * TN + nt) * 16 + lr;
      const float bv = bias ? bias[nlocal] : 0.0f;
#pragma unroll
      for (int r = 0; r < 8; ++r) {
        const int mlocal = (wm * TM + mt) * 16 + r + half * 8;
        float v = acc[mt][nt][r] + bv;
        if (ACT_GELU)
          v = 0.5f * v * (1.0f + erff(v * 0.70710678118654752f));
        if (MASK) {
          const int d = (gi0 + mlocal) - (gj0 + nlocal);
          v = (d >= 0 && d < SPAN) ? v * scale : 0.0f;
        }
        if (OUT_FP32)
          ((float*)Cp)[(long)mlocal * ldc + nlocal] = v;
        else
          ((unsigned short*)Cp)[(long)mlocal * ldc + nlocal] = f2bf(v);
      }
    }
}

// ------------------------------ wrappers -----------------------------------

// out = act(A @ Wt^T + bias); A:[16384][768] bf16, Wt:[768][768] bf16
// 128x128 block: waves 2x4, each wave 4x2 tiles; async HBM->LDS staging
template <int ACT>
__global__ __launch_bounds__(256) void mlp_gemm_kernel(
    const unsigned short* __restrict__ A, const unsigned short* __restrict__ Wt,
    const float* __restrict__ bias, unsigned short* __restrict__ C)
{
  const unsigned short* Ap = A + (size_t)blockIdx.y * 128 * N_EMBD;
  unsigned short* Cp = C + (size_t)blockIdx.y * 128 * N_EMBD + blockIdx.x * 128;
  gemm_block<2, 4, 4, 2, ACT, 0, 0, 0, 1>(
      Ap, N_EMBD, Wt, blockIdx.x * 128, N_EMBD,
      0, (unsigned)N_EMBD, (unsigned)N_EMBD,
      bias + blockIdx.x * 128, Cp, N_EMBD, N_EMBD, 0, 0, 0.0f);
}

// S block [64][320] = band(q[qi0:qi0+64] @ k^T); window j0 = qi0-256
__global__ __launch_bounds__(256) void scores_kernel(
    const unsigned short* __restrict__ q, const unsigned short* __restrict__ k,
    unsigned short* __restrict__ S, float inv_scale)
{
  const int z = blockIdx.z, qb = blockIdx.y, nb = blockIdx.x;
  const int qi0 = qb * 64;
  const int j0b = qi0 - 256 + nb * 64;
  const unsigned short* Ap = q + ((size_t)z * T_LEN + qi0) * N_EMBD;
  const unsigned short* Bt = k + (size_t)z * T_LEN * N_EMBD;
  unsigned short* Cp = S + (((size_t)z * (T_LEN / 64) + qb) * 64) * 320 + nb * 64;
  gemm_block<4, 2, 1, 2, 0, 0, 1, 1, 0>(
      Ap, N_EMBD, Bt, j0b, N_EMBD,
      0, (unsigned)T_LEN, (unsigned)N_EMBD,
      nullptr, Cp, 320, N_EMBD, qi0, j0b, inv_scale);
}

// y block [64][768] = S_block [64][320] @ v_window ; uses vT[d][t]
// 64x128 block: waves 2x4, each wave 2x2 tiles
__global__ __launch_bounds__(256) void attn_y_kernel(
    const unsigned short* __restrict__ S, const unsigned short* __restrict__ vT,
    float* __restrict__ y)
{
  const int z = blockIdx.z, qb = blockIdx.y, nb = blockIdx.x;
  const int qi0 = qb * 64;
  const int j0  = qi0 - 256;
  const unsigned short* Ap = S + (((size_t)z * (T_LEN / 64) + qb) * 64) * 320;
  const unsigned short* Bt = vT + (size_t)z * N_EMBD * T_LEN +
                             (size_t)nb * 128 * T_LEN;
  float* Cp = y + ((size_t)z * T_LEN + qi0) * N_EMBD + nb * 128;
  gemm_block<2, 4, 2, 2, 0, 1, 0, 1, 0>(
      Ap, 320, Bt, 0, T_LEN,
      j0, 128u, (unsigned)T_LEN,
      nullptr, Cp, N_EMBD, 320, 0, 0, 0.0f);
}

// ------------------------- data-movement kernels ---------------------------

__global__ void convert_x_kernel(const float* __restrict__ x,
                                 unsigned short* __restrict__ xb, long n)
{
  for (long i = (long)blockIdx.x * blockDim.x + threadIdx.x; i < n;
       i += (long)gridDim.x * blockDim.x)
    xb[i] = f2bf(x[i]);
}

// W[768][768] fp32 (in,out) -> Wt[768][768] bf16 (out,in)
__global__ void transpose_w_kernel(const float* __restrict__ W,
                                   unsigned short* __restrict__ Wt)
{
  __shared__ float t[32][33];
  const int c0 = blockIdx.x * 32, r0 = blockIdx.y * 32;
  const int tx = threadIdx.x, ty = threadIdx.y;
#pragma unroll
  for (int i = 0; i < 4; ++i)
    t[ty + 8 * i][tx] = W[(size_t)(r0 + ty + 8 * i) * N_EMBD + c0 + tx];
  __syncthreads();
#pragma unroll
  for (int i = 0; i < 4; ++i)
    Wt[(size_t)(c0 + ty + 8 * i) * N_EMBD + r0 + tx] = f2bf(t[tx][ty + 8 * i]);
}

// per-batch v[4096][768] bf16 -> vT[768][4096] bf16
__global__ void transpose_v_kernel(const unsigned short* __restrict__ v,
                                   unsigned short* __restrict__ vT)
{
  __shared__ unsigned short t[32][33];
  const int z = blockIdx.z;
  const int c0 = blockIdx.x * 32, r0 = blockIdx.y * 32;  // r over T, c over D
  const int tx = threadIdx.x, ty = threadIdx.y;
  const unsigned short* src = v + (size_t)z * T_LEN * N_EMBD;
  unsigned short* dst = vT + (size_t)z * N_EMBD * T_LEN;
#pragma unroll
  for (int i = 0; i < 4; ++i)
    t[ty + 8 * i][tx] = src[(size_t)(r0 + ty + 8 * i) * N_EMBD + c0 + tx];
  __syncthreads();
#pragma unroll
  for (int i = 0; i < 4; ++i)
    dst[(size_t)(c0 + ty + 8 * i) * T_LEN + r0 + tx] = t[tx][ty + 8 * i];
}

// LayerNorm over last dim (768), one block per row
__global__ __launch_bounds__(256) void ln_kernel(
    const float* __restrict__ y, const float* __restrict__ w,
    const float* __restrict__ b, float* __restrict__ out)
{
  const size_t row = blockIdx.x;
  const float* yr = y + row * N_EMBD;
  float vals[3], s = 0.0f, s2 = 0.0f;
#pragma unroll
  for (int i = 0; i < 3; ++i) {
    const float v = yr[threadIdx.x + 256 * i];
    vals[i] = v; s += v; s2 += v * v;
  }
  __shared__ float rs[256], rq[256];
  rs[threadIdx.x] = s; rq[threadIdx.x] = s2;
  __syncthreads();
  for (int off = 128; off > 0; off >>= 1) {
    if ((int)threadIdx.x < off) {
      rs[threadIdx.x] += rs[threadIdx.x + off];
      rq[threadIdx.x] += rq[threadIdx.x + off];
    }
    __syncthreads();
  }
  const float mean = rs[0] * (1.0f / N_EMBD);
  const float var  = rq[0] * (1.0f / N_EMBD) - mean * mean;
  const float inv  = rsqrtf(var + 1e-5f);
#pragma unroll
  for (int i = 0; i < 3; ++i) {
    const int c = threadIdx.x + 256 * i;
    out[row * N_EMBD + c] = (vals[i] - mean) * inv * w[c] + b[c];
  }
}

// ------------------------------- launcher ----------------------------------

extern "C" void kernel_launch(void* const* d_in, const int* in_sizes, int n_in,
                              void* d_out, int out_size, void* d_ws, size_t ws_size,
                              hipStream_t stream)
{
  (void)in_sizes; (void)n_in; (void)out_size; (void)ws_size;
  const float* x = (const float*)d_in[0];
  const float* W1[3] = {(const float*)d_in[1], (const float*)d_in[5],  (const float*)d_in[9]};
  const float* B1[3] = {(const float*)d_in[2], (const float*)d_in[6],  (const float*)d_in[10]};
  const float* W2[3] = {(const float*)d_in[3], (const float*)d_in[7],  (const float*)d_in[11]};
  const float* B2[3] = {(const float*)d_in[4], (const float*)d_in[8],  (const float*)d_in[12]};
  const float* ln_w = (const float*)d_in[13];
  const float* ln_b = (const float*)d_in[14];

  char* ws = (char*)d_ws;
  const size_t SZ_ACT = (size_t)BATCH * T_LEN * N_EMBD * 2;   // 25,165,824 B
  unsigned short* xb   = (unsigned short*)(ws);
  unsigned short* hbuf = (unsigned short*)(ws + 1 * SZ_ACT);
  unsigned short* qbuf = (unsigned short*)(ws + 2 * SZ_ACT);
  unsigned short* kbuf = (unsigned short*)(ws + 3 * SZ_ACT);
  unsigned short* vbuf = (unsigned short*)(ws + 4 * SZ_ACT);
  unsigned short* vT   = (unsigned short*)(ws + 5 * SZ_ACT);
  unsigned short* wt   = (unsigned short*)(ws + 6 * SZ_ACT);  // 6x768x768 bf16
  // reuse dead regions:
  unsigned short* S = xb;                       // 10.5 MB <= 25.2 MB (xb dead)
  float* y = (float*)(ws + 2 * SZ_ACT);         // 50.3 MB over q+k (dead)

  const float inv_scale = 1.0f / sqrtf((float)N_EMBD * (float)SPAN);
  const size_t WSZ = (size_t)N_EMBD * N_EMBD;

  // 1) precision conversion / layout
  convert_x_kernel<<<4096, 256, 0, stream>>>(x, xb, (long)BATCH * T_LEN * N_EMBD);
  for (int i = 0; i < 3; ++i) {
    transpose_w_kernel<<<dim3(24, 24), dim3(32, 8), 0, stream>>>(W1[i], wt + (size_t)(2 * i) * WSZ);
    transpose_w_kernel<<<dim3(24, 24), dim3(32, 8), 0, stream>>>(W2[i], wt + (size_t)(2 * i + 1) * WSZ);
  }

  // 2) q/k/v MLPs (WMMA GEMMs, 128x128 blocks, async LDS staging)
  unsigned short* outs[3] = {qbuf, kbuf, vbuf};
  for (int i = 0; i < 3; ++i) {
    mlp_gemm_kernel<1><<<dim3(6, 128), 256, 0, stream>>>(
        xb, wt + (size_t)(2 * i) * WSZ, B1[i], hbuf);
    mlp_gemm_kernel<0><<<dim3(6, 128), 256, 0, stream>>>(
        hbuf, wt + (size_t)(2 * i + 1) * WSZ, B2[i], outs[i]);
  }

  // 3) banded attention
  transpose_v_kernel<<<dim3(24, 128, BATCH), dim3(32, 8), 0, stream>>>(vbuf, vT);
  scores_kernel<<<dim3(5, T_LEN / 64, BATCH), 256, 0, stream>>>(qbuf, kbuf, S, inv_scale);
  attn_y_kernel<<<dim3(6, T_LEN / 64, BATCH), 256, 0, stream>>>(S, vT, y);

  // 4) LayerNorm -> output
  ln_kernel<<<dim3(BATCH * T_LEN), 256, 0, stream>>>(y, ln_w, ln_b, (float*)d_out);
}
